// Mesh2_14267881357853
// MI455X (gfx1250) — matrix operand order repaired and checked by reading, not compiled, not run on hardware
//
#include <hip/hip_runtime.h>

typedef __attribute__((ext_vector_type(16))) __bf16 v16bf;
typedef __attribute__((ext_vector_type(8)))  __bf16 v8bf;
typedef __attribute__((ext_vector_type(4)))  __bf16 v4bf;
typedef __attribute__((ext_vector_type(8)))  float  v8f;
typedef __attribute__((ext_vector_type(4)))  float  v4f;

// Convert 4 f32 -> packed bf16 quad (lowers to v_cvt_pk_bf16_f32 pairs)
__device__ __forceinline__ v4bf cvt4(v4f v) {
    v4bf h;
    h[0] = (__bf16)v[0]; h[1] = (__bf16)v[1];
    h[2] = (__bf16)v[2]; h[3] = (__bf16)v[3];
    return h;
}

// ---------------------------------------------------------------------------
// Weight pre-pack: W [512, K] f32 row-major  ->  fragment-major bf16.
// Fragment id f = nt*KT + kt; lane's 16 B-matrix values stored contiguously:
//   out[(f*32 + lane)*16 + j] = (bf16) W[(nt*16 + (lane&15))*K
//                                        + kt*32 + (lane>>4)*16 + j]
// In the GEMM a B fragment is ONE 32-byte per-lane load, fully coalesced
// (32 lanes x 32B = 1KB contiguous per fragment), no converts in hot loop.
// ---------------------------------------------------------------------------
template <int K>
__global__ __launch_bounds__(256)
void pack_w_kernel(const float* __restrict__ W, __bf16* __restrict__ out) {
    constexpr int KT = K / 32;
    const int id = blockIdx.x * 256 + threadIdx.x;   // 0 .. 32*K-1
    if (id >= 32 * K) return;
    const int lane = id & 31;
    const int frag = id >> 5;
    const int kt   = frag % KT;
    const int nt   = frag / KT;
    const int n    = nt * 16 + (lane & 15);
    const int koff = kt * 32 + (lane >> 4) * 16;
    const float* wp = W + (size_t)n * K + koff;
    v4f w0 = *(const v4f*)(wp);
    v4f w1 = *(const v4f*)(wp + 4);
    v4f w2 = *(const v4f*)(wp + 8);
    v4f w3 = *(const v4f*)(wp + 12);
    __bf16* op = out + (size_t)id * 16;
    *(v4bf*)(op)      = cvt4(w0);
    *(v4bf*)(op + 4)  = cvt4(w1);
    *(v4bf*)(op + 8)  = cvt4(w2);
    *(v4bf*)(op + 12) = cvt4(w3);
}

// ---------------------------------------------------------------------------
// Shared GEMM core: 32 rows of bf16 activations staged in LDS (row stride
// K+8 elements => conflict-free ds_load_b128), times W^T, f32 accumulate via
// v_wmma_f32_16x16x32_bf16. 8 waves: wave>>2 selects 16-row mtile,
// wave&3 selects a group of 8 column tiles (8*4*16 = 512 output cols).
// WPREP=true : pre-packed bf16 B; TWO column tiles interleaved per iteration
//              (independent B streams + accumulators -> loads stay in flight
//              behind the wmmas instead of s_wait_loadcnt 0 per wmma).
// WPREP=false: fallback with on-the-fly f32->bf16 conversion of W.
// ---------------------------------------------------------------------------
template <int K, bool WPREP>
__device__ __forceinline__ void gemm_rows32(const __bf16* __restrict__ Abuf,
                                            const float*  __restrict__ W,
                                            const __bf16* __restrict__ Wp,
                                            const float*  __restrict__ bias,
                                            float*        __restrict__ outp,
                                            int row0blk, int N) {
    constexpr int LDA = K + 8;      // padded row stride (bf16 elements)
    constexpr int KT  = K / 32;     // k-tiles of 32
    const int tid   = threadIdx.x;
    const int w     = tid >> 5;     // wave id 0..7
    const int lane  = tid & 31;
    const int nlane = lane & 15;    // column within 16-wide tile / A-row (M)
    const int half  = lane >> 4;    // which 16-lane half
    const int mtile = w >> 2;       // 0..1  -> rows [mtile*16, +16)
    const int wq    = w & 3;        // 0..3  -> 8 column tiles each
    const int row0  = row0blk + mtile * 16;

    // A fragment base for this lane: M = lane&15, K-16 block selected by half
    // (contiguous k order used here: elements 0..15 = K half*16..+15, matches packed B)
    const __bf16* arow = Abuf + (size_t)(mtile * 16 + nlane) * LDA + half * 16;

    if (WPREP) {
        for (int nt = wq * 8; nt < wq * 8 + 8; nt += 2) {
            // Two adjacent column tiles: their packed streams are contiguous
            // (each stream = KT*512 bf16 = KT KB).
            const __bf16* wf0 = Wp + ((size_t)nt * KT * 32 + lane) * 16;
            const __bf16* wf1 = wf0 + (size_t)KT * 512;
            // warm the following pair's stream (speculative, SE scope)
            __builtin_prefetch(wf1 + (size_t)KT * 512, 0, 3);
            v8f acc0 = {};
            v8f acc1 = {};
#pragma unroll
            for (int kt = 0; kt < KT; ++kt) {
                v16bf bf0 = *(const v16bf*)(wf0 + (size_t)kt * 512);
                v16bf bf1 = *(const v16bf*)(wf1 + (size_t)kt * 512);
                v8bf lo = *(const v8bf*)(arow + kt * 32);
                v8bf hi = *(const v8bf*)(arow + kt * 32 + 8);
                v16bf af = __builtin_shufflevector(lo, hi,
                    0, 1, 2, 3, 4, 5, 6, 7, 8, 9, 10, 11, 12, 13, 14, 15);
                acc0 = __builtin_amdgcn_wmma_f32_16x16x32_bf16(
                    false, af, false, bf0, (short)0, acc0, false, false);
                acc1 = __builtin_amdgcn_wmma_f32_16x16x32_bf16(
                    false, af, false, bf1, (short)0, acc1, false, false);
            }
            // C/D layout: VGPR r -> row (r + 8*half), lane&15 -> column
            const int ncol0 = nt * 16 + nlane;
            const float bv0 = bias[ncol0];
            const float bv1 = bias[ncol0 + 16];
#pragma unroll
            for (int r = 0; r < 8; ++r) {
                const int row = row0 + r + 8 * half;
                if (row < N) {
                    float* orow = outp + (size_t)row * 512 + ncol0;
                    orow[0]  = acc0[r] + bv0;
                    orow[16] = acc1[r] + bv1;
                }
            }
        }
    } else {
        for (int nt = wq * 8; nt < wq * 8 + 8; ++nt) {
            const int ncol = nt * 16 + nlane;
            v8f acc = {};
            const float* wbase = W + (size_t)ncol * K + half * 16;
            __builtin_prefetch(wbase, 0, 3);
#pragma unroll
            for (int kt = 0; kt < KT; ++kt) {
                v8bf lo = *(const v8bf*)(arow + kt * 32);
                v8bf hi = *(const v8bf*)(arow + kt * 32 + 8);
                v16bf af = __builtin_shufflevector(lo, hi,
                    0, 1, 2, 3, 4, 5, 6, 7, 8, 9, 10, 11, 12, 13, 14, 15);
                const float* wp = wbase + kt * 32;
                v4f w0 = *(const v4f*)(wp);
                v4f w1 = *(const v4f*)(wp + 4);
                v4f w2 = *(const v4f*)(wp + 8);
                v4f w3 = *(const v4f*)(wp + 12);
                v16bf bfr;
                bfr[0]  = (__bf16)w0[0]; bfr[1]  = (__bf16)w0[1];
                bfr[2]  = (__bf16)w0[2]; bfr[3]  = (__bf16)w0[3];
                bfr[4]  = (__bf16)w1[0]; bfr[5]  = (__bf16)w1[1];
                bfr[6]  = (__bf16)w1[2]; bfr[7]  = (__bf16)w1[3];
                bfr[8]  = (__bf16)w2[0]; bfr[9]  = (__bf16)w2[1];
                bfr[10] = (__bf16)w2[2]; bfr[11] = (__bf16)w2[3];
                bfr[12] = (__bf16)w3[0]; bfr[13] = (__bf16)w3[1];
                bfr[14] = (__bf16)w3[2]; bfr[15] = (__bf16)w3[3];
                acc = __builtin_amdgcn_wmma_f32_16x16x32_bf16(
                    false, af, false, bfr, (short)0, acc, false, false);
            }
            const float bv = bias[ncol];
#pragma unroll
            for (int r = 0; r < 8; ++r) {
                const int row = row0 + r + 8 * half;
                if (row < N) outp[(size_t)row * 512 + ncol] = acc[r] + bv;
            }
        }
    }
}

// ---------------------------------------------------------------------------
// Kernel 1: out3 = concat(out1,out2) @ W_comb^T + b_comb     (K = 512)
// ---------------------------------------------------------------------------
template <bool WPREP>
__global__ __launch_bounds__(256)
void mesh2_comb_kernel(const float* __restrict__ out1,
                       const float* __restrict__ out2,
                       const float* __restrict__ Wc,
                       const __bf16* __restrict__ Wcp,
                       const float* __restrict__ bc,
                       float* __restrict__ out3, int N) {
    constexpr int K = 512, LDA = K + 8;
    __shared__ __bf16 Abuf[32 * LDA];          // 33,280 B
    const int row0 = blockIdx.x * 32;

    // Stage 32 rows x 512 cols of concat(out1,out2) as bf16.
    // 4096 float4 total; 256 threads x 16 iterations.
#pragma unroll
    for (int i = 0; i < 16; ++i) {
        const int idx = threadIdx.x + i * 256;  // 0..4095
        const int r   = idx >> 7;               // 128 float4 per row
        const int col = (idx & 127) << 2;
        const int rg  = row0 + r;
        v4f v = {};
        if (rg < N) {
            const float* src = (col < 256)
                ? (out1 + (size_t)rg * 256 + col)
                : (out2 + (size_t)rg * 256 + (col - 256));
            v = *(const v4f*)src;
        }
        *(v4bf*)(&Abuf[r * LDA + col]) = cvt4(v);
    }
    __syncthreads();

    gemm_rows32<K, WPREP>(Abuf, Wc, Wcp, bc, out3, row0, N);
}

// ---------------------------------------------------------------------------
// Kernel 2: agg = 0.25*(out2[n] + out2[i0] + out2[i1] + out2[i2]);
//           out4 = agg @ W_agg^T + b_agg                      (K = 256)
// ---------------------------------------------------------------------------
template <bool WPREP>
__global__ __launch_bounds__(256)
void mesh2_agg_kernel(const float* __restrict__ out2,
                      const long long* __restrict__ nbr,
                      const float* __restrict__ Wa,
                      const __bf16* __restrict__ Wap,
                      const float* __restrict__ ba,
                      float* __restrict__ out4, int N) {
    constexpr int K = 256, LDA = K + 8;
    __shared__ __bf16 Abuf[32 * LDA];          // 16,896 B
    const int row0 = blockIdx.x * 32;

    // Gather + mean staging: 8 threads per row, 32 cols (8 float4) each.
    {
        const int r   = threadIdx.x >> 3;      // 0..31
        const int sub = threadIdx.x & 7;
        const int rg  = row0 + r;
        const float *s0 = out2, *s1 = out2, *s2 = out2, *s3 = out2;
        const bool ok = (rg < N);
        if (ok) {
            const long long* nb = nbr + (size_t)rg * 3;
            s0 = out2 + (size_t)rg * 256;
            s1 = out2 + (size_t)nb[0] * 256;
            s2 = out2 + (size_t)nb[1] * 256;
            s3 = out2 + (size_t)nb[2] * 256;
        }
#pragma unroll
        for (int i = 0; i < 8; ++i) {
            const int col = sub * 32 + i * 4;
            v4f v = {};
            if (ok) {
                v4f a = *(const v4f*)(s0 + col);
                v4f b = *(const v4f*)(s1 + col);
                v4f c = *(const v4f*)(s2 + col);
                v4f d = *(const v4f*)(s3 + col);
                v = (a + b + c + d) * 0.25f;
            }
            *(v4bf*)(&Abuf[r * LDA + col]) = cvt4(v);
        }
    }
    __syncthreads();

    gemm_rows32<K, WPREP>(Abuf, Wa, Wap, ba, out4, row0, N);
}

// ---------------------------------------------------------------------------
extern "C" void kernel_launch(void* const* d_in, const int* in_sizes, int n_in,
                              void* d_out, int out_size, void* d_ws, size_t ws_size,
                              hipStream_t stream) {
    const float*     out1 = (const float*)d_in[0];
    const float*     out2 = (const float*)d_in[1];
    const long long* nbr  = (const long long*)d_in[2];   // int64 indices
    const float*     Wc   = (const float*)d_in[3];
    const float*     bc   = (const float*)d_in[4];
    const float*     Wa   = (const float*)d_in[5];
    const float*     ba   = (const float*)d_in[6];

    const int N = in_sizes[0] / 256;
    float* o3 = (float*)d_out;                 // tuple order: (out3, out4)
    float* o4 = o3 + (size_t)N * 512;

    const int grid = (N + 31) / 32;
    const size_t need = (size_t)(512 * 512 + 512 * 256) * 2;   // 786,432 B

    if (ws_size >= need) {
        __bf16* Wcp = (__bf16*)d_ws;              // 512x512 packed bf16
        __bf16* Wap = Wcp + 512 * 512;            // 512x256 packed bf16
        pack_w_kernel<512><<<(32 * 512) / 256, 256, 0, stream>>>(Wc, Wcp);
        pack_w_kernel<256><<<(32 * 256) / 256, 256, 0, stream>>>(Wa, Wap);
        mesh2_comb_kernel<true><<<grid, 256, 0, stream>>>(out1, out2, Wc, Wcp, bc, o3, N);
        mesh2_agg_kernel<true> <<<grid, 256, 0, stream>>>(out2, nbr, Wa, Wap, ba, o4, N);
    } else {
        mesh2_comb_kernel<false><<<grid, 256, 0, stream>>>(out1, out2, Wc, nullptr, bc, o3, N);
        mesh2_agg_kernel<false> <<<grid, 256, 0, stream>>>(out2, nbr, Wa, nullptr, ba, o4, N);
    }
}
